// LSA_EncoderLayer_36730560315528
// MI455X (gfx1250) — compile-verified, matrix operand order checked by reading, not run on hardware
//
#include <hip/hip_runtime.h>
#include <hip/hip_bf16.h>

// ---------------------------------------------------------------------------
// Types
// ---------------------------------------------------------------------------
typedef _Float16 half_t;
typedef __attribute__((ext_vector_type(16))) _Float16 v16h;
typedef __attribute__((ext_vector_type(8)))  _Float16 v8h;
typedef __attribute__((ext_vector_type(8)))  float    v8f;
typedef __attribute__((ext_vector_type(4)))  float    v4f;   // 16-byte copy unit

// Problem constants
#define BATCH 8
#define SEQ   4096
#define DMOD  256
#define NWIN  8
#define WINL  512
#define NH    2
#define KD    256
#define DFF   1024
#define BH    (BATCH * NH)      // 16

__device__ __forceinline__ v16h combine16(v8h lo, v8h hi) {
    v16h r;
#pragma unroll
    for (int i = 0; i < 8; ++i) { r[i] = lo[i]; r[8 + i] = hi[i]; }
    return r;
}

// Build a 16x32 f16 A/B fragment from an LDS tile row (row-major, 32 halfs
// of K per row, 16B-aligned rows).  Per CDNA5 ISA layout: lane half h needs
// K = [h*8 .. h*8+7] in regs 0..7 and K = [16+h*8 ..] in regs 8..15.
template <int STRIDE>
__device__ __forceinline__ v16h frag_ld(const half_t* base, int row, int half_id) {
    const half_t* p = base + (size_t)row * STRIDE;
    v8h lo = *(const v8h*)(p + half_id * 8);
    v8h hi = *(const v8h*)(p + 16 + half_id * 8);
    return combine16(lo, hi);
}

// ---------------------------------------------------------------------------
// Generic WMMA GEMM:  C[M][N] = A[M][K] (f16, row-major) * Bt[N][K] (f16,
// N-major i.e. pre-transposed B), f32 accumulate, templated epilogue.
// Block = 256 threads = 8 waves; block tile 128x128, wave tile 32x64, BK=32.
// ---------------------------------------------------------------------------
template <typename Epi>
__global__ __launch_bounds__(256, 1) void gemm_wmma_kernel(
    const half_t* __restrict__ A, const half_t* __restrict__ Bt,
    int M, int N, int K, Epi epi)
{
    __shared__ __align__(16) half_t As[128][32];
    __shared__ __align__(16) half_t Bs[128][32];

    const int tid  = threadIdx.x;
    const int lane = tid & 31;
    const int wave = tid >> 5;          // 0..7
    const int wm   = wave >> 1;         // 0..3 (M direction)
    const int wn   = wave & 1;          // 0..1 (N direction)
    const int hid  = lane >> 4;         // half-wave id
    const int l16  = lane & 15;
    const int bm   = blockIdx.x * 128;
    const int bn   = blockIdx.y * 128;

    v8f acc[2][4];
#pragma unroll
    for (int i = 0; i < 2; ++i)
#pragma unroll
        for (int j = 0; j < 4; ++j)
#pragma unroll
            for (int v = 0; v < 8; ++v) acc[i][j][v] = 0.0f;

    for (int k0 = 0; k0 < K; k0 += 32) {
        // Cooperative stage: 128 rows x 64B each for A and Bt.
#pragma unroll
        for (int i = tid; i < 128 * 4; i += 256) {
            int r = i >> 2, c = i & 3;
            *(v4f*)&As[r][c * 8] = *(const v4f*)&A[(size_t)(bm + r) * K + k0 + c * 8];
        }
#pragma unroll
        for (int i = tid; i < 128 * 4; i += 256) {
            int r = i >> 2, c = i & 3;
            *(v4f*)&Bs[r][c * 8] = *(const v4f*)&Bt[(size_t)(bn + r) * K + k0 + c * 8];
        }
        if (k0 + 32 < K) {  // global_prefetch_b8 of next tiles
            __builtin_prefetch(&A [(size_t)(bm + (tid >> 1)) * K + k0 + 32], 0, 1);
            __builtin_prefetch(&Bt[(size_t)(bn + (tid >> 1)) * K + k0 + 32], 0, 1);
        }
        __syncthreads();

        v16h af[2], bf[4];
#pragma unroll
        for (int i = 0; i < 2; ++i)
            af[i] = frag_ld<32>(&As[0][0], wm * 32 + i * 16 + l16, hid);
#pragma unroll
        for (int j = 0; j < 4; ++j)
            bf[j] = frag_ld<32>(&Bs[0][0], wn * 64 + j * 16 + l16, hid);

#pragma unroll
        for (int i = 0; i < 2; ++i)
#pragma unroll
            for (int j = 0; j < 4; ++j)
                acc[i][j] = __builtin_amdgcn_wmma_f32_16x16x32_f16(
                    false, af[i], false, bf[j], (short)0, acc[i][j], false, false);
        __syncthreads();
    }

    // Epilogue: lane holds D(m,n) at m = tile +8*hid+v, n = tile + l16.
#pragma unroll
    for (int i = 0; i < 2; ++i)
#pragma unroll
        for (int j = 0; j < 4; ++j)
#pragma unroll
            for (int v = 0; v < 8; ++v) {
                int m = bm + wm * 32 + i * 16 + hid * 8 + v;
                int n = bn + wn * 64 + j * 16 + l16;
                epi(m, n, acc[i][j][v]);
            }
}

// ---------------------------------------------------------------------------
// Epilogues
// ---------------------------------------------------------------------------
struct EpiQKV {        // scatter q [bh][s][c], k [bh][s][c], v^T [bh][w][c][t]
    const float *bq, *bk, *bv;
    half_t *q, *k, *vt;
    __device__ void operator()(int m, int n, float acc) const {
        int b = m >> 12, s = m & 4095;
        int which = n >> 9;             // 0=q 1=k 2=v
        int hk = n & 511;
        int h = hk >> 8, c = hk & 255;
        size_t bh = (size_t)b * NH + h;
        if (which == 0) {
            q[(bh * SEQ + s) * KD + c] = (half_t)(acc + bq[hk]);
        } else if (which == 1) {
            k[(bh * SEQ + s) * KD + c] = (half_t)(acc + bk[hk]);
        } else {
            int w = s >> 9, t = s & 511;
            vt[((bh * NWIN + w) * KD + c) * (size_t)WINL + t] = (half_t)(acc + bv[hk]);
        }
    }
};

struct EpiBiasResF32 {  // out[m][n] = acc + bias[n] + res[m][n]  (f32 out)
    const float* bias; const float* res; float* out; int N;
    __device__ void operator()(int m, int n, float acc) const {
        out[(size_t)m * N + n] = acc + bias[n] + res[(size_t)m * N + n];
    }
};

struct EpiReluF16 {     // out[m][n] = relu(acc + bias[n])  (f16 out)
    const float* bias; half_t* out; int N;
    __device__ void operator()(int m, int n, float acc) const {
        float v = acc + bias[n];
        out[(size_t)m * N + n] = (half_t)(v > 0.0f ? v : 0.0f);
    }
};

// ---------------------------------------------------------------------------
// Fused windowed attention, flash-style streaming over 4 key chunks of 128:
//   per chunk:  S_c = Q[128x256] @ K_c^T   (wave holds 16x128 = 64 VGPRs)
//               online softmax (running row max/sum, rescale ctx)
//               ctx += exp(S_c - m) @ V_c  (wave ctx tile 16x256 = 128 VGPRs)
// Keeps peak live VGPRs ~230 (vs ~400 for the monolithic version, which the
// compiler spilled to scratch inside the WMMA loop).
// ---------------------------------------------------------------------------
__global__ __launch_bounds__(256, 1) void attn_kernel(
    const half_t* __restrict__ q,    // [BH][SEQ][KD]
    const half_t* __restrict__ kbuf, // [BH][SEQ][KD]
    const half_t* __restrict__ vt,   // [BH][NWIN][KD][WINL]
    half_t* __restrict__ ctx)        // [B*SEQ][NH*KD]
{
    const int sb = blockIdx.x;   // 0..3  s-block within window
    const int w  = blockIdx.y;   // 0..7
    const int bh = blockIdx.z;   // 0..15

    __shared__ __align__(16) char smem[26624];
    half_t (*As)[32]  = (half_t(*)[32])(smem);           // phase A: Q tile   8KB
    half_t (*Bs1)[32] = (half_t(*)[32])(smem + 8192);    // phase A: K tile   8KB
    half_t (*Pn)[40]  = (half_t(*)[40])(smem);           // phase B: P stage 10KB
    half_t (*Bs2)[32] = (half_t(*)[32])(smem + 10240);   // phase B: V^T tile 16KB

    const int tid  = threadIdx.x;
    const int lane = tid & 31;
    const int wave = tid >> 5;       // 0..7, owns rows wave*16..+15
    const int hid  = lane >> 4;
    const int l16  = lane & 15;

    const half_t* qbase = q    + ((size_t)bh * SEQ + w * WINL + sb * 128) * KD;
    const half_t* kwin  = kbuf + ((size_t)bh * SEQ + w * WINL) * KD;
    const half_t* vbase = vt   + ((size_t)bh * NWIN + w) * KD * (size_t)WINL;

    // Persistent state: ctx accumulator (16 rows x 256 cols) + row max/sum.
    v8f cacc[16];
#pragma unroll
    for (int j = 0; j < 16; ++j)
#pragma unroll
        for (int v = 0; v < 8; ++v) cacc[j][v] = 0.0f;
    float rmax[8], rsum[8];
#pragma unroll
    for (int v = 0; v < 8; ++v) { rmax[v] = -3.4e38f; rsum[v] = 0.0f; }

    for (int tc = 0; tc < 4; ++tc) {          // 4 key/value chunks of 128
        const half_t* kcbase = kwin + (size_t)(tc * 128) * KD;

        // ---- Phase A: score chunk S = Q @ K_c^T  (wave tile 16 x 128) ----
        v8f sacc[8];
#pragma unroll
        for (int t = 0; t < 8; ++t)
#pragma unroll
            for (int v = 0; v < 8; ++v) sacc[t][v] = 0.0f;

        for (int k0 = 0; k0 < KD; k0 += 32) {
#pragma unroll
            for (int i = tid; i < 128 * 4; i += 256) {
                int r = i >> 2, c = i & 3;
                *(v4f*)&As[r][c * 8] = *(const v4f*)&qbase[(size_t)r * KD + k0 + c * 8];
            }
#pragma unroll
            for (int i = tid; i < 128 * 4; i += 256) {
                int r = i >> 2, c = i & 3;
                *(v4f*)&Bs1[r][c * 8] = *(const v4f*)&kcbase[(size_t)r * KD + k0 + c * 8];
            }
            __syncthreads();
            v16h af = frag_ld<32>(&As[0][0], wave * 16 + l16, hid);
#pragma unroll
            for (int t = 0; t < 8; ++t) {
                v16h bf = frag_ld<32>(&Bs1[0][0], t * 16 + l16, hid);
                sacc[t] = __builtin_amdgcn_wmma_f32_16x16x32_f16(
                    false, af, false, bf, (short)0, sacc[t], false, false);
            }
            __syncthreads();
        }

        // ---- Online softmax update (scale 1/sqrt(256) = 1/16) ----
        float alpha[8];
#pragma unroll
        for (int v = 0; v < 8; ++v) {
            float cm = -3.4e38f;
#pragma unroll
            for (int t = 0; t < 8; ++t) {
                sacc[t][v] *= 0.0625f;
                cm = fmaxf(cm, sacc[t][v]);
            }
#pragma unroll
            for (int mk = 1; mk < 16; mk <<= 1) cm = fmaxf(cm, __shfl_xor(cm, mk, 32));
            float nm = fmaxf(rmax[v], cm);
            float a  = __expf(rmax[v] - nm);   // 0 on first chunk
            float cs = 0.0f;
#pragma unroll
            for (int t = 0; t < 8; ++t) {
                float e = __expf(sacc[t][v] - nm);
                sacc[t][v] = e;
                cs += e;
            }
#pragma unroll
            for (int mk = 1; mk < 16; mk <<= 1) cs += __shfl_xor(cs, mk, 32);
            rsum[v] = rsum[v] * a + cs;
            rmax[v] = nm;
            alpha[v] = a;
        }
#pragma unroll
        for (int j = 0; j < 16; ++j)
#pragma unroll
            for (int v = 0; v < 8; ++v) cacc[j][v] *= alpha[v];

        // ---- Phase B: ctx += P_chunk @ V_chunk  (k = 128, 4 substeps) ----
        for (int kc = 0; kc < 4; ++kc) {
#pragma unroll
            for (int tt = 0; tt < 2; ++tt) {
                int t = kc * 2 + tt;
#pragma unroll
                for (int v = 0; v < 8; ++v) {
                    int mrow = wave * 16 + hid * 8 + v;
                    Pn[mrow][tt * 16 + l16] = (half_t)sacc[t][v];
                }
            }
#pragma unroll
            for (int i = tid; i < 256 * 4; i += 256) {
                int r = i >> 2, c = i & 3;
                *(v4f*)&Bs2[r][c * 8] =
                    *(const v4f*)&vbase[(size_t)r * WINL + tc * 128 + kc * 32 + c * 8];
            }
            __syncthreads();
            v16h af = frag_ld<40>(&Pn[0][0], wave * 16 + l16, hid);
#pragma unroll
            for (int j = 0; j < 16; ++j) {
                v16h bf = frag_ld<32>(&Bs2[0][0], j * 16 + l16, hid);
                cacc[j] = __builtin_amdgcn_wmma_f32_16x16x32_f16(
                    false, af, false, bf, (short)0, cacc[j], false, false);
            }
            __syncthreads();
        }
    }

    // ---- Epilogue: normalize by running sum, write ctx f16 ----
    float inv[8];
#pragma unroll
    for (int v = 0; v < 8; ++v) inv[v] = 1.0f / rsum[v];

    const int b = bh >> 1, h = bh & 1;
    const size_t rowbase = (size_t)b * SEQ + w * WINL + sb * 128;
#pragma unroll
    for (int j = 0; j < 16; ++j)
#pragma unroll
        for (int v = 0; v < 8; ++v) {
            int mloc = wave * 16 + hid * 8 + v;
            int n = h * KD + j * 16 + l16;
            ctx[(rowbase + mloc) * (size_t)(NH * KD) + n] = (half_t)(cacc[j][v] * inv[v]);
        }
}

// ---------------------------------------------------------------------------
// Row-wise LayerNorm over 256 features (one row per 256-thread block).
// Writes f32 (residual carry / final out) and optionally f16 (GEMM operand).
// ---------------------------------------------------------------------------
__global__ __launch_bounds__(256) void ln_kernel(
    const float* __restrict__ in, const float* __restrict__ g,
    const float* __restrict__ beta, float* __restrict__ of32,
    half_t* __restrict__ of16)
{
    const int row = blockIdx.x, tid = threadIdx.x;
    const size_t base = (size_t)row * DMOD;
    float v = in[base + tid];

    __shared__ float red[8];
    float s = v;
#pragma unroll
    for (int m = 1; m < 32; m <<= 1) s += __shfl_xor(s, m, 32);
    if ((tid & 31) == 0) red[tid >> 5] = s;
    __syncthreads();
    float mu = 0.0f;
#pragma unroll
    for (int i = 0; i < 8; ++i) mu += red[i];
    mu *= (1.0f / DMOD);
    __syncthreads();

    float d = v - mu, s2 = d * d;
#pragma unroll
    for (int m = 1; m < 32; m <<= 1) s2 += __shfl_xor(s2, m, 32);
    if ((tid & 31) == 0) red[tid >> 5] = s2;
    __syncthreads();
    float var = 0.0f;
#pragma unroll
    for (int i = 0; i < 8; ++i) var += red[i];
    var *= (1.0f / DMOD);

    float y = d * rsqrtf(var + 1e-3f) * g[tid] + beta[tid];
    of32[base + tid] = y;
    if (of16) of16[base + tid] = (half_t)y;
}

// ---------------------------------------------------------------------------
// Small prep kernels
// ---------------------------------------------------------------------------
__global__ void f32_to_f16_kernel(const float* __restrict__ in,
                                  half_t* __restrict__ out, int n) {
    int i = blockIdx.x * 256 + threadIdx.x;
    if (i < n) out[i] = (half_t)in[i];
}

// out[c][r] = (f16) in[r][c]   (in: R x C f32  ->  out: C x R f16, N-major B)
__global__ void transpose_f16_kernel(const float* __restrict__ in,
                                     half_t* __restrict__ out, int R, int C) {
    int i = blockIdx.x * 256 + threadIdx.x;
    if (i < R * C) {
        int r = i / C, c = i % C;
        out[(size_t)c * R + r] = (half_t)in[i];
    }
}

// ---------------------------------------------------------------------------
// Host-side orchestration
// ---------------------------------------------------------------------------
extern "C" void kernel_launch(void* const* d_in, const int* in_sizes, int n_in,
                              void* d_out, int out_size, void* d_ws, size_t ws_size,
                              hipStream_t stream) {
    (void)in_sizes; (void)n_in; (void)out_size; (void)ws_size;
    const float* x     = (const float*)d_in[0];
    const float* Wq    = (const float*)d_in[1];
    const float* bq    = (const float*)d_in[2];
    const float* Wk    = (const float*)d_in[3];
    const float* bk    = (const float*)d_in[4];
    const float* Wv    = (const float*)d_in[5];
    const float* bv    = (const float*)d_in[6];
    const float* Wo    = (const float*)d_in[7];
    const float* bo    = (const float*)d_in[8];
    const float* ln1_g = (const float*)d_in[9];
    const float* ln1_b = (const float*)d_in[10];
    const float* W1    = (const float*)d_in[11];
    const float* b1    = (const float*)d_in[12];
    const float* W2    = (const float*)d_in[13];
    const float* b2    = (const float*)d_in[14];
    const float* ln2_g = (const float*)d_in[15];
    const float* ln2_b = (const float*)d_in[16];
    float* out = (float*)d_out;

    const int    M      = BATCH * SEQ;                    // 32768
    const size_t SZ_XH  = (size_t)M * DMOD * 2;           // 16 MB (alias y1h)
    const size_t SZ_WQK = 1536ull * DMOD * 2;
    const size_t SZ_WO  = 256ull * 512 * 2;
    const size_t SZ_W1  = 1024ull * DMOD * 2;
    const size_t SZ_W2  = 256ull * DFF * 2;
    const size_t SZ_QKV = (size_t)BH * SEQ * KD * 2;      // 32 MB each
    const size_t SZ_CTX = (size_t)M * (NH * KD) * 2;      // 32 MB
    const size_t SZ_Y32 = (size_t)M * DMOD * 4;           // 32 MB

    char* ws = (char*)d_ws;
    size_t o = 0;
    half_t* xh    = (half_t*)(ws + o); o += SZ_XH;
    half_t* wqkvT = (half_t*)(ws + o); o += SZ_WQK;
    half_t* woT   = (half_t*)(ws + o); o += SZ_WO;
    half_t* w1T   = (half_t*)(ws + o); o += SZ_W1;
    half_t* w2T   = (half_t*)(ws + o); o += SZ_W2;
    size_t oq = o;
    half_t* qb    = (half_t*)(ws + o); o += SZ_QKV;
    half_t* kb    = (half_t*)(ws + o); o += SZ_QKV;
    half_t* vtb   = (half_t*)(ws + o); o += SZ_QKV;
    half_t* ctx   = (half_t*)(ws + o); o += SZ_CTX;
    float*  y0    = (float*)(ws + o);  o += SZ_Y32;       // also y2
    float*  y1f   = (float*)(ws + o);  o += SZ_Y32;
    half_t* hbuf  = (half_t*)(ws + oq);                   // alias q+k (64 MB)
    half_t* y1h   = xh;                                   // alias xh
    float*  y2    = y0;                                   // alias y0

    // ---- prep: f16 conversions / weight transposes (N-major B operands) ----
    f32_to_f16_kernel<<<(M * DMOD + 255) / 256, 256, 0, stream>>>(x, xh, M * DMOD);
    transpose_f16_kernel<<<(256 * 512 + 255) / 256, 256, 0, stream>>>(Wq, wqkvT, 256, 512);
    transpose_f16_kernel<<<(256 * 512 + 255) / 256, 256, 0, stream>>>(Wk, wqkvT + 512 * 256, 256, 512);
    transpose_f16_kernel<<<(256 * 512 + 255) / 256, 256, 0, stream>>>(Wv, wqkvT + 1024 * 256, 256, 512);
    transpose_f16_kernel<<<(512 * 256 + 255) / 256, 256, 0, stream>>>(Wo, woT, 512, 256);
    transpose_f16_kernel<<<(256 * 1024 + 255) / 256, 256, 0, stream>>>(W1, w1T, 256, 1024);
    transpose_f16_kernel<<<(1024 * 256 + 255) / 256, 256, 0, stream>>>(W2, w2T, 1024, 256);

    // ---- QKV projection: [32768 x 256] @ [256 x 1536] ----
    {
        EpiQKV epi{bq, bk, bv, qb, kb, vtb};
        gemm_wmma_kernel<EpiQKV><<<dim3(M / 128, 1536 / 128), 256, 0, stream>>>(
            xh, wqkvT, M, 1536, DMOD, epi);
    }

    // ---- windowed attention (flash-style, fused softmax) ----
    attn_kernel<<<dim3(WINL / 128, NWIN, BH), 256, 0, stream>>>(qb, kb, vtb, ctx);

    // ---- out-proj + residual: y0 = ctx @ Wo + bo + x ----
    {
        EpiBiasResF32 epi{bo, x, y0, DMOD};
        gemm_wmma_kernel<EpiBiasResF32><<<dim3(M / 128, DMOD / 128), 256, 0, stream>>>(
            ctx, woT, M, DMOD, NH * KD, epi);
    }

    // ---- LN1 -> y1 (f32 residual carry) + y1h (f16 GEMM operand) ----
    ln_kernel<<<M, DMOD, 0, stream>>>(y0, ln1_g, ln1_b, y1f, y1h);

    // ---- FFN1: relu(y1 @ W1 + b1) -> hbuf (f16) ----
    {
        EpiReluF16 epi{b1, hbuf, DFF};
        gemm_wmma_kernel<EpiReluF16><<<dim3(M / 128, DFF / 128), 256, 0, stream>>>(
            y1h, w1T, M, DFF, DMOD, epi);
    }

    // ---- FFN2 + residual: y2 = hbuf @ W2 + b2 + y1 ----
    {
        EpiBiasResF32 epi{b2, y1f, y2, DMOD};
        gemm_wmma_kernel<EpiBiasResF32><<<dim3(M / 128, DMOD / 128), 256, 0, stream>>>(
            hbuf, w2T, M, DMOD, DFF, epi);
    }

    // ---- LN2 -> final output (f32) ----
    ln_kernel<<<M, DMOD, 0, stream>>>(y2, ln2_g, ln2_b, out, (half_t*)nullptr);
}
